// RAWTextureDetector_88244398063693
// MI455X (gfx1250) — matrix operand order repaired and checked by reading, not compiled, not run on hardware
//
#include <hip/hip_runtime.h>
#include <hip/hip_bf16.h>

typedef _Float16 f16;
typedef _Float16 v16h __attribute__((ext_vector_type(16)));
typedef _Float16 v8h  __attribute__((ext_vector_type(8)));
typedef float    v8f  __attribute__((ext_vector_type(8)));
typedef unsigned int u32x4 __attribute__((ext_vector_type(4)));
typedef int          i32x8 __attribute__((ext_vector_type(8)));
typedef int          i32x4 __attribute__((ext_vector_type(4)));

#define HDIM 1536
#define WDIM 1536
#define HW   (HDIM*WDIM)
#define NBINS 4096

// ---- workspace layout (bytes), regions aliased in time ----
#define OFF_F1     0ull
#define OFF_SUMH   OFF_F1
#define OFF_SUMSQH (OFF_F1 + (size_t)8*HW*4)
#define OFF_F2     150994944ull
#define OFF_MAPS   OFF_F2
#define OFF_FUSED  (OFF_F2 + (size_t)3*HW*4)
#define OFF_F3     226492416ull
#define OFF_WA1    301989888ull                        // 32*64 f16
#define OFF_WA2    (OFF_WA1 + 4096)                    // 16*288 f16
#define OFF_WA3    (OFF_WA2 + 9216)                    // 16*160 f16
#define OFF_STAT   (OFF_WA3 + 5120)                    // 8 f32
#define OFF_MM     (OFF_STAT + 32)                     // 2 u32
#define OFF_TH     (OFF_MM + 8)                        // 4 f32
#define OFF_HIST   (OFF_TH + 16)                       // 4096 u32

// ======================= init =======================
__global__ __launch_bounds__(256) void init_k(float* stats, unsigned int* mm, unsigned int* hist) {
    int t = blockIdx.x * 256 + threadIdx.x;
    if (t < 8) stats[t] = 0.f;
    if (t == 8) mm[0] = 0x7f800000u;
    if (t == 9) mm[1] = 0u;
    if (t < NBINS) hist[t] = 0u;
}

// ======================= weight prep: f32 -> f16, permuted K = dx*QP + (ic*3+dy) =======================
__global__ __launch_bounds__(256) void prep_w(const float* __restrict__ w1, const float* __restrict__ w2,
                                              const float* __restrict__ w3,
                                              f16* __restrict__ wA1, f16* __restrict__ wA2, f16* __restrict__ wA3) {
    int t = blockIdx.x * 256 + threadIdx.x;
    if (t < 32 * 64) {                       // conv1: CIN=4, QP=16, KPAD=64
        int oc = t >> 6, kp = t & 63;
        int dx = kp >> 4, q = kp & 15;
        wA1[t] = (dx < 3 && q < 12) ? (f16)w1[((oc * 4 + q / 3) * 3 + q % 3) * 3 + dx] : (f16)0.f;
    }
    if (t < 16 * 288) {                      // conv2: CIN=32, QP=96, KPAD=288 (all real)
        int oc = t / 288, kp = t % 288;
        int dx = kp / 96, q = kp % 96;
        wA2[t] = (f16)w2[((oc * 32 + q / 3) * 3 + q % 3) * 3 + dx];
    }
    if (t < 16 * 160) {                      // conv3: CIN=16, QP=48, KPAD=160, oc padded to 16
        int oc = t / 160, kp = t % 160;
        int dx = kp / 48, q = kp % 48;
        wA3[t] = (oc < 8 && dx < 3) ? (f16)w3[((oc * 16 + q / 3) * 3 + q % 3) * 3 + dx] : (f16)0.f;
    }
}

// ======================= conv3x3 via WMMA (implicit GEMM, TDM-staged input) =======================
// K-permutation: k' = dx*QP + q, q = ic*3+dy, QP multiple of 16 so every 16-aligned K group of a
// lane lies in one dx block -> B fragment is ONE contiguous v16h LDS load from tileC[(n+dx)*QP+q0].
// Positions with dx==3 (K pad) or q>=CIN*3 carry zero weights; tileC pad regions zeroed for NaN safety.
template<int CIN, int COUT, int KPAD, int QP, bool IN_HALF, bool OUT_HALF, bool DO_LEAKY>
__global__ __launch_bounds__(128)
void conv3x3_wmma(const void* __restrict__ inp, const f16* __restrict__ wA,
                  const float* __restrict__ bias, void* __restrict__ outp) {
    constexpr int MT = (COUT > 16) ? 2 : 1;
    constexpr int NT = 4 / MT;
    constexpr int NPIX = NT * 16;
    constexpr int TW = NPIX + 2;
    constexpr int TW2 = ((KPAD > 3 * QP) ? (TW + 2) : TW);  // extra cols for virtual dx=3 block
    constexpr int NCHUNK = KPAD / 32;
    constexpr int ESZ = IN_HALF ? 2 : 4;

    __shared__ alignas(16) unsigned char tileRawB[CIN * 3 * TW * ESZ]; // [ic][dy][col], TDM order
    __shared__ alignas(16) f16 tileC[TW2 * QP];                        // [col][q]

    const int tid = threadIdx.x;
    const int lane = tid & 31;
    const int wv = tid >> 5;
    const int mt = wv % MT;
    const int nt = wv / MT;

    const int pblk = blockIdx.x * NPIX;     // NPIX divides WDIM -> row segment never wraps
    const int y0 = pblk / WDIM;
    const int xb = pblk % WDIM;

    // ---- stage raw tile: TDM for interior blocks, manual guarded loads at left/top edges ----
    const bool interior = (y0 > 0) && (xb > 0);
    if (interior) {
        if (wv == 0) {
            unsigned lds_off = (unsigned)(size_t)(const void*)tileRawB;
            unsigned long long ga = (unsigned long long)(size_t)inp +
                                    (unsigned long long)ESZ * ((size_t)(y0 - 1) * WDIM + (xb - 1));
            unsigned T0 = (unsigned)(WDIM - (xb - 1));   // remaining extent dim0 (cols)
            unsigned T1 = (unsigned)(HDIM - (y0 - 1));   // remaining extent dim1 (rows)
            unsigned long long S0 = (unsigned long long)WDIM;  // row stride (elements)
            unsigned long long S1 = (unsigned long long)HW;    // channel stride (elements)
            u32x4 g0; i32x8 g1; i32x4 g2, g3; i32x8 g4;
            g0[0] = 1u;                                          // count=1, user mode
            g0[1] = lds_off;                                     // lds_addr
            g0[2] = (unsigned)(ga & 0xffffffffu);                // global_addr[31:0]
            g0[3] = (unsigned)((ga >> 32) & 0x01ffffffu) | (2u << 30);  // addr[56:32] | type=2
            g1[0] = (int)((unsigned)(IN_HALF ? 1u : 2u) << 16);  // data_size (1=2B, 2=4B)
            g1[1] = (int)((T0 & 0xffffu) << 16);                 // tensor_dim0 lo16 @ [31:16]
            g1[2] = (int)((T0 >> 16) | ((T1 & 0xffffu) << 16));  // dim0 hi | dim1 lo
            g1[3] = (int)((T1 >> 16) | ((unsigned)TW << 16));    // dim1 hi | tile_dim0
            g1[4] = (int)(3u | ((unsigned)CIN << 16));           // tile_dim1=3 | tile_dim2=CIN
            g1[5] = (int)(S0 & 0xffffffffu);                     // dim0_stride lo32
            g1[6] = (int)((unsigned)(S0 >> 32) | (unsigned)((S1 & 0xffffu) << 16)); // dim0 hi16 | dim1 lo16
            g1[7] = (int)(S1 >> 16);                             // dim1_stride hi32
            g2[0] = CIN; g2[1] = 0; g2[2] = 0; g2[3] = 0;        // tensor_dim2 = CIN
            g3[0] = 0; g3[1] = 0; g3[2] = 0; g3[3] = 0;
            g4[0] = 0; g4[1] = 0; g4[2] = 0; g4[3] = 0;
            g4[4] = 0; g4[5] = 0; g4[6] = 0; g4[7] = 0;
            __builtin_amdgcn_tensor_load_to_lds(g0, g1, g2, g3, g4, 0);
            __builtin_amdgcn_s_wait_tensorcnt(0);
        }
    } else {
        for (int idx = tid; idx < CIN * 3 * TW; idx += 128) {
            int q = idx / TW;              // ic*3+dy
            int col = idx % TW;
            int gy = y0 + (q % 3) - 1;
            int gx = xb + col - 1;
            bool ok = (gy >= 0 && gy < HDIM && gx >= 0 && gx < WDIM);
            size_t off = (size_t)(q / 3) * HW + (size_t)gy * WDIM + gx;
            if constexpr (IN_HALF) ((f16*)tileRawB)[idx]   = ok ? ((const f16*)inp)[off]   : (f16)0.f;
            else                   ((float*)tileRawB)[idx] = ok ? ((const float*)inp)[off] : 0.f;
        }
    }

    // zero K-pad regions of tileC (multiplied by zero weights; must be finite)
    if constexpr (TW2 > TW) {
        for (int idx = tid; idx < (TW2 - TW) * QP; idx += 128) tileC[TW * QP + idx] = (f16)0.f;
    }
    if constexpr (QP > CIN * 3) {
        constexpr int QPAD = QP - CIN * 3;
        for (int idx = tid; idx < TW * QPAD; idx += 128) {
            int col = idx / QPAD, q = CIN * 3 + idx % QPAD;
            tileC[col * QP + q] = (f16)0.f;
        }
    }
    __syncthreads();

    // ---- permute raw tile -> column-major patch layout tileC[col][q] (+ f32->f16 for conv1) ----
    for (int idx = tid; idx < CIN * 3 * TW; idx += 128) {
        int q = idx / TW;
        int col = idx % TW;
        f16 v;
        if constexpr (IN_HALF) v = ((const f16*)tileRawB)[idx];
        else                   v = (f16)((const float*)tileRawB)[idx];
        tileC[col * QP + q] = v;
    }
    __syncthreads();

    // ---- WMMA main loop ----
    const int mrow = mt * 16 + (lane & 15);
    const int kbA = 16 * (lane >> 4);
    const int ncol = nt * 16 + (lane & 15);  // tile-local pixel index
    const int kbB = 16 * (lane >> 4);

    v8f c = {};
    #pragma unroll
    for (int ch = 0; ch < NCHUNK; ++ch) {
        union { v16h v; v8h h[2]; } a;
        const f16* wp = wA + (size_t)mrow * KPAD + ch * 32 + kbA;
        a.h[0] = *(const v8h*)(wp);
        a.h[1] = *(const v8h*)(wp + 8);

        const int G = ch * 32 + kbB;
        const int dx = G / QP;
        const int q0 = G - dx * QP;
        v16h b = *(const v16h*)(tileC + (ncol + dx) * QP + q0);

        c = __builtin_amdgcn_wmma_f32_16x16x32_f16(false, a.v, false, b, (short)0, c, false, false);
    }

    // C/D layout: lane l, VGPR r -> M = r + 8*(l>>4), N = l&15
    #pragma unroll
    for (int r = 0; r < 8; ++r) {
        int mr = r + 8 * (lane >> 4);
        if (COUT < 16 && mr >= COUT) continue;
        int m = mt * 16 + mr;
        float val = c[r] + bias[m];
        if (DO_LEAKY) val = (val >= 0.f) ? val : 0.2f * val;
        size_t o = (size_t)m * HW + (size_t)pblk + nt * 16 + (lane & 15);
        if constexpr (OUT_HALF) ((f16*)outp)[o] = (f16)val;
        else                    ((float*)outp)[o] = val;
    }
}

// ======================= separable box filter (reflect pad) =======================
__device__ __forceinline__ int reflect_i(int i, int n) {
    i = (i < 0) ? -i : i;
    i = (i >= n) ? (2 * n - 2 - i) : i;
    return i;
}

template<int WS>
__global__ __launch_bounds__(256)
void box_h(const float* __restrict__ f, float* __restrict__ sh, float* __restrict__ sq) {
    constexpr int hw = WS / 2;
    size_t idx = (size_t)blockIdx.x * 256 + threadIdx.x;   // over 8*HW
    int c = (int)(idx / HW);
    int rem = (int)(idx % HW);
    int y = rem / WDIM;
    int x = rem % WDIM;
    const float* row = f + (size_t)c * HW + (size_t)y * WDIM;
    __builtin_prefetch(row + (x + hw + 1 < WDIM ? x + hw + 1 : x), 0, 1);
    float s = 0.f, s2 = 0.f;
    for (int i = -hw; i <= hw; ++i) {
        int xi = x + i;
        xi = (xi < 0) ? -xi : xi;
        xi = (xi >= WDIM) ? (2 * WDIM - 2 - xi) : xi;
        float v = row[xi];
        s += v; s2 += v * v;
    }
    sh[idx] = s; sq[idx] = s2;
}

template<int WS>
__global__ __launch_bounds__(256)
void box_v(const float* __restrict__ sh, const float* __restrict__ sq, float* __restrict__ map) {
    constexpr int hw = WS / 2;
    constexpr float inv = 1.f / (float)(WS * WS);
    const int SEG = 128;
    int tid = blockIdx.x * 256 + threadIdx.x;   // W * (H/SEG) threads
    int x = tid % WDIM;
    int y0 = (tid / WDIM) * SEG;
    float s[8], s2[8];
    #pragma unroll
    for (int c = 0; c < 8; ++c) { s[c] = 0.f; s2[c] = 0.f; }
    for (int i = -hw; i <= hw; ++i) {
        int ry = reflect_i(y0 + i, HDIM);
        #pragma unroll
        for (int c = 0; c < 8; ++c) {
            size_t o = (size_t)c * HW + (size_t)ry * WDIM + x;
            s[c] += sh[o]; s2[c] += sq[o];
        }
    }
    for (int y = y0; y < y0 + SEG; ++y) {
        float acc = 0.f;
        #pragma unroll
        for (int c = 0; c < 8; ++c) {
            float mean = s[c] * inv;
            float msq = s2[c] * inv;
            float var = msq - mean * mean;
            var = (var < 1e-6f) ? 1e-6f : var;
            acc += sqrtf(var);
        }
        map[(size_t)y * WDIM + x] = powf(acc * 0.125f, 0.8f);
        int ya = reflect_i(y + 1 + hw, HDIM);
        int ys = reflect_i(y - hw, HDIM);
        #pragma unroll
        for (int c = 0; c < 8; ++c) {
            size_t oa = (size_t)c * HW + (size_t)ya * WDIM + x;
            size_t os = (size_t)c * HW + (size_t)ys * WDIM + x;
            s[c]  += sh[oa] - sh[os];
            s2[c] += sq[oa] - sq[os];
        }
    }
}

// ======================= per-channel x stats =======================
__global__ __launch_bounds__(256)
void xstats_k(const float* __restrict__ x, float* __restrict__ stats) {
    int c = blockIdx.y;
    const float* p = x + (size_t)c * HW;
    float s = 0.f, s2 = 0.f;
    for (size_t i = (size_t)blockIdx.x * 256 + threadIdx.x; i < (size_t)HW; i += (size_t)gridDim.x * 256) {
        float v = p[i]; s += v; s2 += v * v;
    }
    __shared__ float sh[256], sh2[256];
    sh[threadIdx.x] = s; sh2[threadIdx.x] = s2;
    __syncthreads();
    for (int off = 128; off > 0; off >>= 1) {
        if (threadIdx.x < off) { sh[threadIdx.x] += sh[threadIdx.x + off]; sh2[threadIdx.x] += sh2[threadIdx.x + off]; }
        __syncthreads();
    }
    if (threadIdx.x == 0) { atomicAdd(&stats[c * 2], sh[0]); atomicAdd(&stats[c * 2 + 1], sh2[0]); }
}

// ======================= fuse maps + min/max =======================
__global__ __launch_bounds__(256)
void fuse_k(const float* __restrict__ maps, const float* __restrict__ wf, const float* __restrict__ bf,
            float* __restrict__ fused, unsigned int* __restrict__ mm) {
    size_t p = (size_t)blockIdx.x * 256 + threadIdx.x;
    float v = wf[0] * maps[p] + wf[1] * maps[(size_t)HW + p] + wf[2] * maps[2 * (size_t)HW + p] + bf[0];
    v = (v > 0.f) ? v : 0.f;
    fused[p] = v;
    __shared__ unsigned int smin, smax;
    if (threadIdx.x == 0) { smin = 0x7f800000u; smax = 0u; }
    __syncthreads();
    unsigned int b = __float_as_uint(v);   // v >= 0, bit pattern monotonic
    atomicMin(&smin, b); atomicMax(&smax, b);
    __syncthreads();
    if (threadIdx.x == 0) { atomicMin(&mm[0], smin); atomicMax(&mm[1], smax); }
}

// ======================= histogram of fused =======================
__global__ __launch_bounds__(256)
void hist_k(const float* __restrict__ fused, const unsigned int* __restrict__ mm, unsigned int* __restrict__ hist) {
    __shared__ unsigned int h[NBINS];
    for (int i = threadIdx.x; i < NBINS; i += 256) h[i] = 0u;
    __syncthreads();
    float vmin = __uint_as_float(mm[0]);
    float vmax = __uint_as_float(mm[1]);
    float scale = (vmax > vmin) ? (float)NBINS / (vmax - vmin) : 0.f;
    size_t p = (size_t)blockIdx.x * 256 + threadIdx.x;
    int b = (int)((fused[p] - vmin) * scale);
    b = (b < 0) ? 0 : ((b > NBINS - 1) ? NBINS - 1 : b);
    atomicAdd(&h[b], 1u);
    __syncthreads();
    for (int i = threadIdx.x; i < NBINS; i += 256) if (h[i]) atomicAdd(&hist[i], h[i]);
}

// ======================= scalar thresholds =======================
__global__ void thresh_k(const float* __restrict__ stats, const float* __restrict__ cwraw,
                         const unsigned int* __restrict__ mm, const unsigned int* __restrict__ hist,
                         float* __restrict__ th) {
    float mx = cwraw[0];
    for (int c = 1; c < 4; ++c) mx = fmaxf(mx, cwraw[c]);
    float e[4], es = 0.f;
    for (int c = 0; c < 4; ++c) { e[c] = expf(cwraw[c] - mx); es += e[c]; }
    float lsum = 0.f, usum = 0.f;
    for (int c = 0; c < 4; ++c) {
        float sum = stats[2 * c], sq = stats[2 * c + 1];
        float var = (sq - sum * sum / (float)HW) / (float)(HW - 1);
        float gstd = sqrtf(fmaxf(var, 0.f));
        float gfac = fminf(fmaxf(gstd * 5.f, 0.5f), 2.f);
        float cstd = (e[c] / es) * gstd;                 // std(x*cw) = cw*std(x)
        float cfac = fminf(fmaxf(cstd * 2.f, 0.8f), 1.2f);
        lsum += 0.05f * gfac * cfac;
        usum += 0.20f * gfac * cfac;
    }
    float lmean = lsum * 0.25f, umean = usum * 0.25f;
    float vmin = __uint_as_float(mm[0]);
    float vmax = __uint_as_float(mm[1]);
    float binw = (vmax > vmin) ? (vmax - vmin) / (float)NBINS : 0.f;
    double t30 = 0.3 * (double)(HW - 1), t70 = 0.7 * (double)(HW - 1);
    float q30 = vmin, q70 = vmin;
    double cum = 0.0;
    for (int i = 0; i < NBINS; ++i) {
        double cnt = (double)hist[i];
        if (cum <= t30 && cum + cnt > t30)
            q30 = vmin + ((float)i + (cnt > 0.0 ? (float)((t30 - cum) / cnt) : 0.f)) * binw;
        if (cum <= t70 && cum + cnt > t70)
            q70 = vmin + ((float)i + (cnt > 0.0 ? (float)((t70 - cum) / cnt) : 0.f)) * binw;
        cum += cnt;
    }
    float lth = 0.5f * (lmean + q30);
    float uth = 0.5f * (umean + q70);
    if (uth - lth < 0.01f) uth = lth + 0.01f;
    th[0] = lth;
    th[1] = 1.f / (uth - lth);
}

// ======================= segmented sigmoid output =======================
__global__ __launch_bounds__(256)
void final_k(const float* __restrict__ fused, const float* __restrict__ th, float* __restrict__ out) {
    size_t p = (size_t)blockIdx.x * 256 + threadIdx.x;
    float ns = (fused[p] - th[0]) * th[1];
    ns = (ns < 0.f) ? 0.f : ((ns > 1.f) ? 1.f : ns);
    float r;
    if (ns < 0.25f)      r = 0.4f / (1.f + expf(-(10.f * ns - 1.5f)));
    else if (ns <= 0.6f) r = 0.5f / (1.f + expf(-(20.f * ns - 8.f))) + 0.25f;
    else                 r = 0.3f / (1.f + expf(-(8.f * ns - 1.f))) + 0.7f;
    out[p] = r;
}

// ======================= host launch =======================
extern "C" void kernel_launch(void* const* d_in, const int* in_sizes, int n_in,
                              void* d_out, int out_size, void* d_ws, size_t ws_size,
                              hipStream_t stream) {
    const float* x  = (const float*)d_in[0];
    const float* w1 = (const float*)d_in[1];
    const float* b1 = (const float*)d_in[2];
    const float* w2 = (const float*)d_in[3];
    const float* b2 = (const float*)d_in[4];
    const float* w3 = (const float*)d_in[5];
    const float* b3 = (const float*)d_in[6];
    const float* wf = (const float*)d_in[7];
    const float* bf = (const float*)d_in[8];
    const float* cw = (const float*)d_in[9];
    float* out = (float*)d_out;

    char* ws = (char*)d_ws;
    f16*   f1h   = (f16*)(ws + OFF_F1);
    f16*   f2h   = (f16*)(ws + OFF_F2);
    float* f3    = (float*)(ws + OFF_F3);
    float* sumH  = (float*)(ws + OFF_SUMH);
    float* sumSQ = (float*)(ws + OFF_SUMSQH);
    float* maps  = (float*)(ws + OFF_MAPS);
    float* fused = (float*)(ws + OFF_FUSED);
    f16*   wA1   = (f16*)(ws + OFF_WA1);
    f16*   wA2   = (f16*)(ws + OFF_WA2);
    f16*   wA3   = (f16*)(ws + OFF_WA3);
    float* stats = (float*)(ws + OFF_STAT);
    unsigned int* mm   = (unsigned int*)(ws + OFF_MM);
    float* th    = (float*)(ws + OFF_TH);
    unsigned int* hist = (unsigned int*)(ws + OFF_HIST);

    init_k<<<16, 256, 0, stream>>>(stats, mm, hist);
    prep_w<<<18, 256, 0, stream>>>(w1, w2, w3, wA1, wA2, wA3);

    // convs via WMMA:  <CIN, COUT, KPAD, QP, IN_HALF, OUT_HALF, LEAKY>
    conv3x3_wmma<4, 32, 64, 16, false, true, true><<<HW / 32, 128, 0, stream>>>(x, wA1, b1, f1h);
    conv3x3_wmma<32, 16, 288, 96, true, true, true><<<HW / 64, 128, 0, stream>>>(f1h, wA2, b2, f2h);
    conv3x3_wmma<16, 8, 160, 48, true, false, false><<<HW / 64, 128, 0, stream>>>(f2h, wA3, b3, f3);

    // multi-scale std maps (separable box filters, reflect pad); sumH/sumSQ alias f1h region
    const int bhGrid = 8 * HW / 256;              // 73728
    const int bvGrid = WDIM * (HDIM / 128) / 256; // 72
    box_h<11><<<bhGrid, 256, 0, stream>>>(f3, sumH, sumSQ);
    box_v<11><<<bvGrid, 256, 0, stream>>>(sumH, sumSQ, maps + 0 * (size_t)HW);
    box_h<25><<<bhGrid, 256, 0, stream>>>(f3, sumH, sumSQ);
    box_v<25><<<bvGrid, 256, 0, stream>>>(sumH, sumSQ, maps + 1 * (size_t)HW);
    box_h<49><<<bhGrid, 256, 0, stream>>>(f3, sumH, sumSQ);
    box_v<49><<<bvGrid, 256, 0, stream>>>(sumH, sumSQ, maps + 2 * (size_t)HW);

    // global stats + fusion + quantiles + output
    xstats_k<<<dim3(1024, 4), 256, 0, stream>>>(x, stats);
    fuse_k<<<HW / 256, 256, 0, stream>>>(maps, wf, bf, fused, mm);
    hist_k<<<HW / 256, 256, 0, stream>>>(fused, mm, hist);
    thresh_k<<<1, 1, 0, stream>>>(stats, cw, mm, hist, th);
    final_k<<<HW / 256, 256, 0, stream>>>(fused, th, out);

    (void)in_sizes; (void)n_in; (void)out_size; (void)ws_size;
}
